// DecoderLayer_72327249264859
// MI455X (gfx1250) — compile-verified, matrix operand order checked by reading, not run on hardware
//
#include <hip/hip_runtime.h>
#include <hip/hip_bf16.h>
#include <cstdint>
#include <cstddef>

// ---------------------------------------------------------------------------
// Types for CDNA5 WMMA (wave32): v_wmma_f32_16x16x32_bf16
// ---------------------------------------------------------------------------
typedef __bf16 bf16;
typedef __bf16 bf16x16 __attribute__((ext_vector_type(16)));
typedef float  floatx8 __attribute__((ext_vector_type(8)));
typedef int    i32x4   __attribute__((ext_vector_type(4)));

union FragAB {
    bf16x16 v;
    uint4   q[2];
};
union Pack8 {
    uint4 q;
    bf16  h[8];
};

__device__ __forceinline__ floatx8 zero8() {
    floatx8 z = {0.f, 0.f, 0.f, 0.f, 0.f, 0.f, 0.f, 0.f};
    return z;
}

// ---------------------------------------------------------------------------
// gfx1250 async global->LDS copy (ASYNCcnt-tracked DMA), with sync fallback.
// Builtin signature (probe-confirmed by diagnostics): (int4 AS1*, int4 AS3*,
// imm offset, imm cpol).
// ---------------------------------------------------------------------------
#if defined(__has_builtin)
#  if __has_builtin(__builtin_amdgcn_global_load_async_to_lds_b128)
#    define HAVE_ASYNC_LDS 1
#  endif
#endif
#ifndef HAVE_ASYNC_LDS
#define HAVE_ASYNC_LDS 0
#endif

#if HAVE_ASYNC_LDS
typedef __attribute__((address_space(1))) i32x4 gl_i32x4;
typedef __attribute__((address_space(3))) i32x4 lds_i32x4;
#endif

__device__ __forceinline__ void async_copy_b128(void* lds_dst, const void* gsrc) {
#if HAVE_ASYNC_LDS
    __builtin_amdgcn_global_load_async_to_lds_b128(
        (gl_i32x4*)(void*)gsrc, (lds_i32x4*)lds_dst, /*offset=*/0, /*cpol=*/0);
#else
    *(uint4*)lds_dst = *(const uint4*)gsrc;
#endif
}

__device__ __forceinline__ void async_wait_all() {
#if HAVE_ASYNC_LDS
#  if __has_builtin(__builtin_amdgcn_s_wait_asynccnt)
    __builtin_amdgcn_s_wait_asynccnt(0);
#  else
    asm volatile("s_wait_asynccnt 0" ::: "memory");
#  endif
#endif
}

// A/B fragment loader for 16x32 bf16 tile stored row-major in LDS.
// ISA layout (16-bit A-matrix 16x32): lane L<16 -> row L, K {0..7,16..23};
// lane L+16 -> row L, K {8..15,24..31}. B-operand uses the identical pattern
// when the matrix is stored as [N][K] (i.e. transposed weights).
__device__ __forceinline__ bf16x16 load_frag(const bf16* tile, int stride) {
    const int lane = threadIdx.x & 31;
    const int row  = lane & 15;
    const int koff = (lane >> 4) << 3;   // 0 or 8
    const bf16* p  = tile + row * stride + koff;
    FragAB f;
    f.q[0] = *(const uint4*)(p);
    f.q[1] = *(const uint4*)(p + 16);
    return f.v;
}

__device__ __forceinline__ floatx8 wmma_bf16(bf16x16 a, bf16x16 b, floatx8 c) {
    return __builtin_amdgcn_wmma_f32_16x16x32_bf16(
        /*neg_a=*/false, a, /*neg_b=*/false, b,
        /*c_mod=*/(short)0, c, /*reuse_a=*/false, /*reuse_b=*/false);
}

// ---------------------------------------------------------------------------
// GEMM: C[M,N] = act(alpha * (A[M,K] @ W[K,N] + bias[N]))
// A bf16 row-major; Wt bf16 stored TRANSPOSED [N][K]; out f32 and/or bf16.
// Block tile 128x256, K-step 32, 256 threads = 8 waves (2x4 wave grid),
// 64x64 per wave = 16 WMMA per K-step. Double-buffered async LDS staging:
// next K-tile DMA is issued before computing the current one.
// ---------------------------------------------------------------------------
#define GBM 128
#define GBN 256
#define GBK 32
#define GLD 40   // padded LDS stride (elements); 80B rows keep 16B alignment

__global__ __launch_bounds__(256)
void gemm_bf16_wmma(const bf16* __restrict__ A, const bf16* __restrict__ Wt,
                    const float* __restrict__ bias, float alpha, int relu,
                    float* __restrict__ Cf, bf16* __restrict__ Cb,
                    int M, int N, int K)
{
    __shared__ __align__(16) bf16 sA[2][GBM][GLD];
    __shared__ __align__(16) bf16 sB[2][GBN][GLD];

    const int tid  = threadIdx.x;
    const int wave = tid >> 5;
    const int bm   = blockIdx.y * GBM;
    const int bn   = blockIdx.x * GBN;
    const int wm   = (wave & 1) * 64;   // 4 m-tiles of 16
    const int wn   = (wave >> 1) * 64;  // 4 n-tiles of 16

    auto stage = [&](int buf, int k0) {
#pragma unroll
        for (int it = 0; it < 2; ++it) {
            int s = tid + it * 256;
            int row = s >> 2;
            int cg  = (s & 3) * 8;
            async_copy_b128(&sA[buf][row][cg],
                            A + (size_t)(bm + row) * K + k0 + cg);
        }
#pragma unroll
        for (int it = 0; it < 4; ++it) {
            int s = tid + it * 256;
            int row = s >> 2;
            int cg  = (s & 3) * 8;
            async_copy_b128(&sB[buf][row][cg],
                            Wt + (size_t)(bn + row) * K + k0 + cg);
        }
    };

    floatx8 acc[4][4];
#pragma unroll
    for (int i = 0; i < 4; ++i)
#pragma unroll
        for (int j = 0; j < 4; ++j) acc[i][j] = zero8();

    stage(0, 0);
    async_wait_all();
    __syncthreads();

    int cur = 0;
    for (int k0 = 0; k0 < K; k0 += GBK) {
        const int nxt = cur ^ 1;
        if (k0 + GBK < K) stage(nxt, k0 + GBK);   // overlap DMA with WMMA

        bf16x16 af[4];
#pragma unroll
        for (int i = 0; i < 4; ++i)
            af[i] = load_frag(&sA[cur][wm + i * 16][0], GLD);
#pragma unroll
        for (int j = 0; j < 4; ++j) {
            bf16x16 bfr = load_frag(&sB[cur][wn + j * 16][0], GLD);
#pragma unroll
            for (int i = 0; i < 4; ++i)
                acc[i][j] = wmma_bf16(af[i], bfr, acc[i][j]);
        }

        async_wait_all();
        __syncthreads();
        cur = nxt;
    }

    // Epilogue: C tile layout -> lane holds col (lane&15), rows rb..rb+7.
    const int lane = tid & 31;
    const int col  = lane & 15;
    const int rb   = (lane >> 4) * 8;
#pragma unroll
    for (int i = 0; i < 4; ++i)
#pragma unroll
        for (int j = 0; j < 4; ++j) {
            const int gn = bn + wn + j * 16 + col;
            const float bv = bias[gn];
#pragma unroll
            for (int r = 0; r < 8; ++r) {
                const size_t gm = (size_t)(bm + wm + i * 16 + rb + r);
                float v = alpha * (acc[i][j][r] + bv);
                if (relu) v = fmaxf(v, 0.0f);
                if (Cf) Cf[gm * N + gn] = v;
                if (Cb) Cb[gm * N + gn] = (bf16)v;
            }
        }
}

// ---------------------------------------------------------------------------
// Flash attention (online softmax), one (b, h, 64-query tile) per block.
// Q/K/V bf16 [B, T|S, 1024] with head h at column offset h*64. Q pre-scaled.
// Key blocks of 64; scores via WMMA -> LDS f32; softmax stats in registers of
// 64 row-owner threads; probs bf16 -> WMMA PV (V staged transposed).
// Q/K staged through the async global->LDS DMA path.
// ---------------------------------------------------------------------------
#define FQ  64
#define FK  64
#define FD  64
#define SQK 72   // bf16 stride for sQ/sK   (dh 64 + 8)
#define SVT 72   // bf16 stride for sVt     (keys 64 + 8)
#define SST 68   // f32 stride for scores/PV
#define SPT 72   // bf16 stride for probs

__global__ __launch_bounds__(256)
void flash_attn_wmma(const bf16* __restrict__ Qm, const bf16* __restrict__ Km,
                     const bf16* __restrict__ Vm, const float* __restrict__ mask,
                     bf16* __restrict__ Out, int Tq, int Skv)
{
    __shared__ __align__(16) bf16  sQ[FQ][SQK];
    __shared__ __align__(16) bf16  sK[FK][SQK];
    __shared__ __align__(16) bf16  sVt[FD][SVT];
    __shared__ __align__(16) float sS[FQ][SST];
    __shared__ __align__(16) bf16  sP[FQ][SPT];
    __shared__ float sFac[FQ];
    __shared__ float sL[FQ];
    float* sPV = &sS[0][0];   // PV result overlays scores (stride SST)

    const int tid  = threadIdx.x;
    const int wave = tid >> 5;
    const int qb   = blockIdx.x * FQ;
    const int h    = blockIdx.y;
    const int b    = blockIdx.z;

    // Stage Q tile once (async DMA): 64 rows x 64 cols bf16.
#pragma unroll
    for (int it = 0; it < 2; ++it) {
        int s   = tid + it * 256;
        int row = s >> 3;
        int cg  = (s & 7) * 8;
        async_copy_b128(&sQ[row][cg],
                        Qm + ((size_t)(b * Tq + qb + row)) * 1024 + h * 64 + cg);
    }

    float m = -1e30f, l = 0.0f;       // row stats (valid for tid < FQ)
    const int orow = tid >> 2;        // output row owned by this thread
    const int ocol = (tid & 3) * 16;  // 16-column segment of dh
    float o[16];
#pragma unroll
    for (int e = 0; e < 16; ++e) o[e] = 0.0f;

    for (int kb = 0; kb < Skv; kb += FK) {
        __syncthreads();  // protect sK/sVt/sPV/sFac from previous iteration

        // Stage K block [64][64] via async DMA.
#pragma unroll
        for (int it = 0; it < 2; ++it) {
            int s   = tid + it * 256;
            int row = s >> 3;
            int cg  = (s & 7) * 8;
            async_copy_b128(&sK[row][cg],
                            Km + ((size_t)(b * Skv + kb + row)) * 1024 + h * 64 + cg);
        }
        // Stage V transposed: sVt[d][key] (element scatter, DScnt path).
#pragma unroll
        for (int it = 0; it < 2; ++it) {
            int s  = tid + it * 256;
            int kk = s >> 3;
            int cg = (s & 7) * 8;
            Pack8 p;
            p.q = *(const uint4*)(Vm + ((size_t)(b * Skv + kb + kk)) * 1024 + h * 64 + cg);
#pragma unroll
            for (int e = 0; e < 8; ++e) sVt[cg + e][kk] = p.h[e];
        }
        async_wait_all();
        __syncthreads();

        // Scores: 16 tiles (4m x 4n); wave -> mt = wave&3, nt = (wave>>2)*2+jn.
        {
            const int mt   = wave & 3;
            const int lane = tid & 31;
            const int col  = lane & 15;
            const int rb   = (lane >> 4) * 8;
#pragma unroll
            for (int jn = 0; jn < 2; ++jn) {
                const int nt = (wave >> 2) * 2 + jn;
                floatx8 a = zero8();
#pragma unroll
                for (int kk = 0; kk < FD; kk += 32) {
                    bf16x16 qa = load_frag(&sQ[mt * 16][kk], SQK);
                    bf16x16 kf = load_frag(&sK[nt * 16][kk], SQK);
                    a = wmma_bf16(qa, kf, a);
                }
                const float* mp = mask +
                    ((size_t)(b * Tq + qb + mt * 16 + rb)) * Skv + kb + nt * 16 + col;
#pragma unroll
                for (int r = 0; r < 8; ++r)
                    sS[mt * 16 + rb + r][nt * 16 + col] = a[r] + mp[(size_t)r * Skv];
            }
        }
        __syncthreads();

        // Online softmax per row.
        if (tid < FQ) {
            float bmax = -1e30f;
#pragma unroll 8
            for (int i = 0; i < FK; ++i) bmax = fmaxf(bmax, sS[tid][i]);
            const float mn  = fmaxf(m, bmax);
            const float fac = __expf(m - mn);
            float sum = 0.0f;
#pragma unroll 8
            for (int i = 0; i < FK; ++i) {
                float p = __expf(sS[tid][i] - mn);
                sP[tid][i] = (bf16)p;
                sum += p;
            }
            l = l * fac + sum;
            m = mn;
            sFac[tid] = fac;
        }
        __syncthreads();

        // PV: 16 tiles (4m x 4n over dh); same wave mapping.
        {
            const int mt   = wave & 3;
            const int lane = tid & 31;
            const int col  = lane & 15;
            const int rb   = (lane >> 4) * 8;
#pragma unroll
            for (int jn = 0; jn < 2; ++jn) {
                const int nt = (wave >> 2) * 2 + jn;
                floatx8 a = zero8();
#pragma unroll
                for (int kk = 0; kk < FK; kk += 32) {
                    bf16x16 pa = load_frag(&sP[mt * 16][kk], SPT);
                    bf16x16 vb = load_frag(&sVt[nt * 16][kk], SVT);
                    a = wmma_bf16(pa, vb, a);
                }
#pragma unroll
                for (int r = 0; r < 8; ++r)
                    sPV[(mt * 16 + rb + r) * SST + nt * 16 + col] = a[r];
            }
        }
        __syncthreads();

        // Rescale-accumulate output registers.
        {
            const float fac = sFac[orow];
#pragma unroll
            for (int e = 0; e < 16; ++e)
                o[e] = o[e] * fac + sPV[orow * SST + ocol + e];
        }
    }

    if (tid < FQ) sL[tid] = l;
    __syncthreads();
    const float linv = 1.0f / sL[orow];
#pragma unroll
    for (int e = 0; e < 16; ++e)
        Out[((size_t)(b * Tq + qb + orow)) * 1024 + h * 64 + ocol + e] =
            (bf16)(o[e] * linv);
}

// ---------------------------------------------------------------------------
// LayerNorm over D=1024: one block (256 thr) per row; optional residual add;
// writes f32 and optionally bf16 copy for the next GEMM.
// ---------------------------------------------------------------------------
__global__ __launch_bounds__(256)
void ln_kernel(const float* __restrict__ x, const float* __restrict__ res,
               const float* __restrict__ g, const float* __restrict__ beta,
               float* __restrict__ outf, bf16* __restrict__ outb)
{
    __shared__ float red[256];
    const int row  = blockIdx.x;
    const int tid  = threadIdx.x;
    const int base = tid * 4;

    float4 v = *(const float4*)(x + (size_t)row * 1024 + base);
    if (res) {
        float4 rv = *(const float4*)(res + (size_t)row * 1024 + base);
        v.x += rv.x; v.y += rv.y; v.z += rv.z; v.w += rv.w;
    }
    red[tid] = v.x + v.y + v.z + v.w;
    __syncthreads();
    for (int st = 128; st > 0; st >>= 1) {
        if (tid < st) red[tid] += red[tid + st];
        __syncthreads();
    }
    const float mu = red[0] * (1.0f / 1024.0f);
    __syncthreads();
    const float dx = v.x - mu, dy = v.y - mu, dz = v.z - mu, dw = v.w - mu;
    red[tid] = dx * dx + dy * dy + dz * dz + dw * dw;
    __syncthreads();
    for (int st = 128; st > 0; st >>= 1) {
        if (tid < st) red[tid] += red[tid + st];
        __syncthreads();
    }
    const float rstd = rsqrtf(red[0] * (1.0f / 1024.0f) + 1e-5f);
    const float4 gv = *(const float4*)(g + base);
    const float4 bv = *(const float4*)(beta + base);
    float4 y;
    y.x = dx * rstd * gv.x + bv.x;
    y.y = dy * rstd * gv.y + bv.y;
    y.z = dz * rstd * gv.z + bv.z;
    y.w = dw * rstd * gv.w + bv.w;
    *(float4*)(outf + (size_t)row * 1024 + base) = y;
    if (outb) {
        bf16* ob = outb + (size_t)row * 1024 + base;
        ob[0] = (bf16)y.x; ob[1] = (bf16)y.y; ob[2] = (bf16)y.z; ob[3] = (bf16)y.w;
    }
}

// h2 = a + b + c  (f32 + bf16 copies)
__global__ void add3_kernel(const float* __restrict__ a, const float* __restrict__ b2,
                            const float* __restrict__ c, float* __restrict__ outf,
                            bf16* __restrict__ outb, int n4)
{
    int i = blockIdx.x * blockDim.x + threadIdx.x;
    if (i >= n4) return;
    float4 av = ((const float4*)a)[i];
    float4 bv = ((const float4*)b2)[i];
    float4 cv = ((const float4*)c)[i];
    float4 r;
    r.x = av.x + bv.x + cv.x;
    r.y = av.y + bv.y + cv.y;
    r.z = av.z + bv.z + cv.z;
    r.w = av.w + bv.w + cv.w;
    ((float4*)outf)[i] = r;
    bf16* ob = outb + (size_t)i * 4;
    ob[0] = (bf16)r.x; ob[1] = (bf16)r.y; ob[2] = (bf16)r.z; ob[3] = (bf16)r.w;
}

__global__ void cvt_kernel(const float* __restrict__ in, bf16* __restrict__ out, int n4)
{
    int i = blockIdx.x * blockDim.x + threadIdx.x;
    if (i >= n4) return;
    float4 v = ((const float4*)in)[i];
    bf16* ob = out + (size_t)i * 4;
    ob[0] = (bf16)v.x; ob[1] = (bf16)v.y; ob[2] = (bf16)v.z; ob[3] = (bf16)v.w;
}

// W[K][N] f32 -> Wt[N][K] bf16 (tiled transpose, once per launch).
__global__ __launch_bounds__(256)
void transpose_cvt_kernel(const float* __restrict__ in, bf16* __restrict__ out,
                          int K, int N)
{
    __shared__ float t[32][33];
    const int n0 = blockIdx.x * 32;
    const int k0 = blockIdx.y * 32;
    const int tx = threadIdx.x & 31;
    const int ty = threadIdx.x >> 5;
#pragma unroll
    for (int i = 0; i < 32; i += 8)
        t[ty + i][tx] = in[(size_t)(k0 + ty + i) * N + n0 + tx];
    __syncthreads();
#pragma unroll
    for (int i = 0; i < 32; i += 8)
        out[(size_t)(n0 + ty + i) * K + k0 + tx] = (bf16)t[tx][ty + i];
}

// ---------------------------------------------------------------------------
// Host orchestration
// ---------------------------------------------------------------------------
static inline void launch_gemm(const bf16* A, const bf16* Wt, const float* bias,
                               float alpha, int relu, float* Cf, bf16* Cb,
                               int M, int N, int K, hipStream_t s)
{
    gemm_bf16_wmma<<<dim3(N / GBN, M / GBM), dim3(256), 0, s>>>(
        A, Wt, bias, alpha, relu, Cf, Cb, M, N, K);
}

extern "C" void kernel_launch(void* const* d_in, const int* in_sizes, int n_in,
                              void* d_out, int out_size, void* d_ws, size_t ws_size,
                              hipStream_t stream)
{
    (void)in_sizes; (void)n_in; (void)out_size; (void)ws_size;

    const float* hidden  = (const float*)d_in[0];
    const float* enc     = (const float*)d_in[1];
    const float* graph   = (const float*)d_in[2];
    const float* m_self  = (const float*)d_in[3];
    const float* m_cross = (const float*)d_in[4];
    const float* m_graph = (const float*)d_in[5];
    const float* W_q     = (const float*)d_in[6];
    const float* b_q     = (const float*)d_in[7];
    const float* W_k     = (const float*)d_in[8];
    const float* b_k     = (const float*)d_in[9];
    const float* W_v     = (const float*)d_in[10];
    const float* b_v     = (const float*)d_in[11];
    const float* W_o     = (const float*)d_in[12];
    const float* b_o     = (const float*)d_in[13];
    const float* ln_g    = (const float*)d_in[14];
    const float* ln_b    = (const float*)d_in[15];
    const float* fc1_w   = (const float*)d_in[16];
    const float* fc1_b   = (const float*)d_in[17];
    const float* fc2_w   = (const float*)d_in[18];
    const float* fc2_b   = (const float*)d_in[19];

    constexpr int Dm = 1024, NHh = 16, Bb = 4, Tt = 1024, Ss = 1024, D4 = 4096;
    constexpr int NT = Bb * Tt;            // 4096 decoder tokens
    constexpr float SC = 0.125f;           // (D/NH)^-0.5 = 64^-0.5

    // Bump-allocate scratch (~224 MB total; harness workspace assumed ample).
    size_t off = 0;
    auto alloc = [&](size_t bytes) -> void* {
        void* p = (char*)d_ws + off;
        off += (bytes + 255) & ~(size_t)255;
        return p;
    };
    bf16*  wqT    = (bf16*)alloc((size_t)3 * Dm * Dm * 2);
    bf16*  wkT    = (bf16*)alloc((size_t)3 * Dm * Dm * 2);
    bf16*  wvT    = (bf16*)alloc((size_t)3 * Dm * Dm * 2);
    bf16*  woT    = (bf16*)alloc((size_t)3 * Dm * Dm * 2);
    bf16*  fc1T   = (bf16*)alloc((size_t)Dm * D4 * 2);
    bf16*  fc2T   = (bf16*)alloc((size_t)Dm * D4 * 2);
    bf16*  xb     = (bf16*)alloc((size_t)NT * Dm * 2);
    bf16*  encb   = (bf16*)alloc((size_t)NT * Dm * 2);
    bf16*  graphb = (bf16*)alloc((size_t)NT * Dm * 2);
    bf16*  qbuf   = (bf16*)alloc((size_t)NT * Dm * 2);
    bf16*  kbuf   = (bf16*)alloc((size_t)NT * Dm * 2);
    bf16*  vbuf   = (bf16*)alloc((size_t)NT * Dm * 2);
    bf16*  abuf   = (bf16*)alloc((size_t)NT * Dm * 2);
    float* tmpf   = (float*)alloc((size_t)NT * Dm * 4);
    float* h1f    = (float*)alloc((size_t)NT * Dm * 4);
    bf16*  h1b    = (bf16*)alloc((size_t)NT * Dm * 2);
    float* encof  = (float*)alloc((size_t)NT * Dm * 4);
    float* graphof= (float*)alloc((size_t)NT * Dm * 4);
    float* h2f    = (float*)alloc((size_t)NT * Dm * 4);
    bf16*  h2b    = (bf16*)alloc((size_t)NT * Dm * 2);
    bf16*  midb   = (bf16*)alloc((size_t)NT * D4 * 2);

    const dim3 T256(256);

    // ---- prep: weight transpose+convert, activation convert ----
    for (int i = 0; i < 3; ++i) {
        const size_t o2 = (size_t)i * Dm * Dm;
        transpose_cvt_kernel<<<dim3(Dm/32, Dm/32), T256, 0, stream>>>(W_q + o2, wqT + o2, Dm, Dm);
        transpose_cvt_kernel<<<dim3(Dm/32, Dm/32), T256, 0, stream>>>(W_k + o2, wkT + o2, Dm, Dm);
        transpose_cvt_kernel<<<dim3(Dm/32, Dm/32), T256, 0, stream>>>(W_v + o2, wvT + o2, Dm, Dm);
        transpose_cvt_kernel<<<dim3(Dm/32, Dm/32), T256, 0, stream>>>(W_o + o2, woT + o2, Dm, Dm);
    }
    transpose_cvt_kernel<<<dim3(D4/32, Dm/32), T256, 0, stream>>>(fc1_w, fc1T, Dm, D4);
    transpose_cvt_kernel<<<dim3(Dm/32, D4/32), T256, 0, stream>>>(fc2_w, fc2T, D4, Dm);

    const int n4 = NT * Dm / 4;
    cvt_kernel<<<n4 / 256, T256, 0, stream>>>(hidden, xb, n4);
    cvt_kernel<<<n4 / 256, T256, 0, stream>>>(enc, encb, n4);
    cvt_kernel<<<n4 / 256, T256, 0, stream>>>(graph, graphb, n4);

    const dim3 fgrid(Tt / FQ, NHh, Bb);

    // ---- stage 1: self attention ----
    launch_gemm(xb, wqT, b_q, SC,  0, nullptr, qbuf, NT, Dm, Dm, stream);
    launch_gemm(xb, wkT, b_k, 1.f, 0, nullptr, kbuf, NT, Dm, Dm, stream);
    launch_gemm(xb, wvT, b_v, 1.f, 0, nullptr, vbuf, NT, Dm, Dm, stream);
    flash_attn_wmma<<<fgrid, T256, 0, stream>>>(qbuf, kbuf, vbuf, m_self, abuf, Tt, Tt);
    launch_gemm(abuf, woT, b_o, 1.f, 0, tmpf, nullptr, NT, Dm, Dm, stream);
    ln_kernel<<<NT, T256, 0, stream>>>(tmpf, hidden, ln_g, ln_b, h1f, h1b);

    // ---- stage 2a: cross attention ----
    launch_gemm(h1b,  wqT + (size_t)1*Dm*Dm, b_q + Dm, SC,  0, nullptr, qbuf, NT, Dm, Dm, stream);
    launch_gemm(encb, wkT + (size_t)1*Dm*Dm, b_k + Dm, 1.f, 0, nullptr, kbuf, NT, Dm, Dm, stream);
    launch_gemm(encb, wvT + (size_t)1*Dm*Dm, b_v + Dm, 1.f, 0, nullptr, vbuf, NT, Dm, Dm, stream);
    flash_attn_wmma<<<fgrid, T256, 0, stream>>>(qbuf, kbuf, vbuf, m_cross, abuf, Tt, Ss);
    launch_gemm(abuf, woT + (size_t)1*Dm*Dm, b_o + Dm, 1.f, 0, tmpf, nullptr, NT, Dm, Dm, stream);
    ln_kernel<<<NT, T256, 0, stream>>>(tmpf, nullptr, ln_g + Dm, ln_b + Dm, encof, nullptr);

    // ---- stage 2b: graph attention ----
    launch_gemm(h1b,    wqT + (size_t)2*Dm*Dm, b_q + 2*Dm, SC,  0, nullptr, qbuf, NT, Dm, Dm, stream);
    launch_gemm(graphb, wkT + (size_t)2*Dm*Dm, b_k + 2*Dm, 1.f, 0, nullptr, kbuf, NT, Dm, Dm, stream);
    launch_gemm(graphb, wvT + (size_t)2*Dm*Dm, b_v + 2*Dm, 1.f, 0, nullptr, vbuf, NT, Dm, Dm, stream);
    flash_attn_wmma<<<fgrid, T256, 0, stream>>>(qbuf, kbuf, vbuf, m_graph, abuf, Tt, Ss);
    launch_gemm(abuf, woT + (size_t)2*Dm*Dm, b_o + 2*Dm, 1.f, 0, tmpf, nullptr, NT, Dm, Dm, stream);
    ln_kernel<<<NT, T256, 0, stream>>>(tmpf, nullptr, ln_g + 2*Dm, ln_b + 2*Dm, graphof, nullptr);

    add3_kernel<<<n4 / 256, T256, 0, stream>>>(h1f, encof, graphof, h2f, h2b, n4);

    // ---- stage 3: FFN ----
    launch_gemm(h2b,  fc1T, fc1_b, 1.f, 1, nullptr, midb, NT, D4, Dm, stream);
    launch_gemm(midb, fc2T, fc2_b, 1.f, 0, tmpf, nullptr, NT, Dm, D4, stream);
    ln_kernel<<<NT, T256, 0, stream>>>(tmpf, h2f, ln_g + 3*Dm, ln_b + 3*Dm,
                                       (float*)d_out, nullptr);
}
